// SpaTranslator_Aligner_28406913695828
// MI455X (gfx1250) — compile-verified, matrix operand order checked by reading, not run on hardware
//
#include <hip/hip_runtime.h>
#include <hip/hip_bf16.h>
#include <math.h>

#define D_IN   512
#define D_LAT  256
#define D_EMB  64
#define NEG_SLOPE 0.2f

typedef __bf16 bf16_t;
typedef bf16_t v16bf __attribute__((ext_vector_type(16)));
typedef bf16_t v8bf  __attribute__((ext_vector_type(8)));
typedef float  v8f   __attribute__((ext_vector_type(8)));

// ---- bit casts via unions (no reliance on HIP intrinsic names)
static __device__ __forceinline__ unsigned f_as_u(float f) {
  union { float f; unsigned u; } v; v.f = f; return v.u;
}
static __device__ __forceinline__ float u_as_f(unsigned u) {
  union { unsigned u; float f; } v; v.u = u; return v.f;
}

// ---- fp32 -> bf16 round-to-nearest-even
static __device__ __forceinline__ bf16_t f2bf(float f) {
  unsigned u = f_as_u(f);
  unsigned r = u + 0x7fffu + ((u >> 16) & 1u);
  union { unsigned short s; bf16_t h; } o; o.s = (unsigned short)(r >> 16);
  return o.h;
}

// ---------------------------------------------------------------------------
// WMMA GEMM on pre-converted bf16 operands.
//   A : [M,K] row-major bf16, leading dim lda
//   Bt: [N,K] row-major bf16 (logical B column n is Bt row n -> contiguous)
//   C : [M,N] row-major fp32
// One 16x16 tile per wave (tile id wave-uniform => EXEC all-ones at WMMA).
// Inner loop: 4x global_load_b128 + 1x v_wmma_f32_16x16x32_bf16.
// ---------------------------------------------------------------------------
__global__ void k_gemm_wmma(const bf16_t* __restrict__ A, const bf16_t* __restrict__ Bt,
                            float* __restrict__ C, int M, int N, int K,
                            int lda, int ldb, int ldc) {
  const int wave  = threadIdx.x >> 5;
  const int lane  = threadIdx.x & 31;
  const int half_ = lane >> 4;     // K-half selector per ISA 16-bit fragment layout
  const int r     = lane & 15;
  const int ntiles = N >> 4;
  const int mtiles = M >> 4;
  int t = blockIdx.x * 8 + wave;
  if (t >= mtiles * ntiles) return;            // whole-wave uniform exit
  const int tm = t / ntiles, tn = t % ntiles;

  // lane-owned pointers: A row (tm*16+r), B column (tn*16+r), K-offset half_*8
  const bf16_t* ap = A  + (size_t)(tm * 16 + r) * lda + half_ * 8;
  const bf16_t* bp = Bt + (size_t)(tn * 16 + r) * ldb + half_ * 8;

  v8f acc = {};
  for (int kb = 0; kb < K; kb += 32) {
    // fragment K-runs: [kb+8h, kb+8h+7] and [kb+16+8h, kb+16+8h+7], each 16B
    v8bf alo = *(const v8bf*)(ap + kb);
    v8bf ahi = *(const v8bf*)(ap + kb + 16);
    v8bf blo = *(const v8bf*)(bp + kb);
    v8bf bhi = *(const v8bf*)(bp + kb + 16);
    v16bf a = __builtin_shufflevector(alo, ahi, 0,1,2,3,4,5,6,7,8,9,10,11,12,13,14,15);
    v16bf b = __builtin_shufflevector(blo, bhi, 0,1,2,3,4,5,6,7,8,9,10,11,12,13,14,15);
    acc = __builtin_amdgcn_wmma_f32_16x16x32_bf16(false, a, false, b,
                                                  (short)0, acc, false, false);
  }
  // C/D layout: lane column = tn*16+r, VGPR i -> row tm*16 + 8*half_ + i
  float* cp = C + (size_t)(tm * 16 + half_ * 8) * ldc + tn * 16 + r;
#pragma unroll
  for (int i = 0; i < 8; ++i) cp[(size_t)i * ldc] = acc[i];
}

// ---------------------------------------------------------------------------
// Streaming fp32 -> bf16 convert (8 elements/thread, 16B stores). n8 = n/8.
// ---------------------------------------------------------------------------
__global__ void k_f32_to_bf16(const float* __restrict__ in, bf16_t* __restrict__ out,
                              long long n8) {
  long long i = (long long)blockIdx.x * blockDim.x + threadIdx.x;
  if (i >= n8) return;
  const float4* p = (const float4*)in + i * 2;
  float4 a = p[0], b = p[1];
  v8bf o;
  o[0] = f2bf(a.x); o[1] = f2bf(a.y); o[2] = f2bf(a.z); o[3] = f2bf(a.w);
  o[4] = f2bf(b.x); o[5] = f2bf(b.y); o[6] = f2bf(b.z); o[7] = f2bf(b.w);
  *(v8bf*)(out + i * 8) = o;
}

// transpose + convert: in[R,C] fp32 -> out[C,R] bf16 (small weight matrices)
__global__ void k_f32_to_bf16_T(const float* __restrict__ in, bf16_t* __restrict__ out,
                                int R, int C) {
  int idx = blockIdx.x * blockDim.x + threadIdx.x;
  if (idx >= R * C) return;
  int c = idx / R, r = idx - c * R;
  out[(size_t)c * R + r] = f2bf(in[(size_t)r * C + c]);
}

// fused ELU + fp32->bf16 (GEMM input producer). n8 = n/8.
__global__ void k_elu_bf16(const float* __restrict__ in, bf16_t* __restrict__ out,
                           long long n8) {
  long long i = (long long)blockIdx.x * blockDim.x + threadIdx.x;
  if (i >= n8) return;
  const float4* p = (const float4*)in + i * 2;
  float4 a = p[0], b = p[1];
  float e[8] = {a.x, a.y, a.z, a.w, b.x, b.y, b.z, b.w};
  v8bf o;
#pragma unroll
  for (int j = 0; j < 8; ++j) {
    float x = e[j];
    o[j] = f2bf(x > 0.f ? x : expm1f(x));
  }
  *(v8bf*)(out + i * 8) = o;
}

// ---------------------------------------------------------------------------
// Per-node attention logits: als[i] = h1[i,:].a_src ; ald[i] = h1[i,:].a_dst
// ---------------------------------------------------------------------------
__global__ void k_logits(const float* __restrict__ h, const float* __restrict__ a_src,
                         const float* __restrict__ a_dst, float* __restrict__ als,
                         float* __restrict__ ald, int NN) {
  int i = blockIdx.x * blockDim.x + threadIdx.x;
  if (i >= NN) return;
  const float4* hp = (const float4*)(h + (size_t)i * D_LAT);
  const float4* as = (const float4*)a_src;
  const float4* ad = (const float4*)a_dst;
  float s1 = 0.f, s2 = 0.f;
#pragma unroll 4
  for (int j = 0; j < D_LAT / 4; ++j) {
    float4 hv = hp[j], sv = as[j], dv = ad[j];
    s1 += hv.x * sv.x + hv.y * sv.y + hv.z * sv.z + hv.w * sv.w;
    s2 += hv.x * dv.x + hv.y * dv.y + hv.z * dv.z + hv.w * dv.w;
  }
  als[i] = s1; ald[i] = s2;
}

__global__ void k_fill_u32(unsigned* __restrict__ p, unsigned v, int n) {
  int i = blockIdx.x * blockDim.x + threadIdx.x;
  if (i < n) p[i] = v;
}

// order-preserving float<->uint for atomic max over possibly-negative logits
static __device__ __forceinline__ unsigned fenc(float f) {
  unsigned u = f_as_u(f);
  return (u & 0x80000000u) ? ~u : (u | 0x80000000u);
}
static __device__ __forceinline__ float fdec(unsigned u) {
  u = (u & 0x80000000u) ? (u & 0x7fffffffu) : ~u;
  return u_as_f(u);
}

static __device__ __forceinline__ float lrelu(float v) {
  return v > 0.f ? v : NEG_SLOPE * v;
}

__global__ void k_edge_max(const int* __restrict__ src, const int* __restrict__ dst,
                           const float* __restrict__ als, const float* __restrict__ ald,
                           unsigned* __restrict__ menc, int E) {
  int e = blockIdx.x * blockDim.x + threadIdx.x;
  if (e >= E) return;
  int d = dst[e];
  float v = lrelu(als[src[e]] + ald[d]);
  atomicMax(&menc[d], fenc(v));
}

__global__ void k_edge_expsum(const int* __restrict__ src, const int* __restrict__ dst,
                              const float* __restrict__ als, const float* __restrict__ ald,
                              const unsigned* __restrict__ menc, float* __restrict__ w,
                              float* __restrict__ ssum, int E) {
  int e = blockIdx.x * blockDim.x + threadIdx.x;
  if (e >= E) return;
  int d = dst[e];
  float v = lrelu(als[src[e]] + ald[d]);
  float ww = expf(v - fdec(menc[d]));
  w[e] = ww;
  atomicAdd(&ssum[d], ww);
}

// one wave per edge: 32 lanes x 8 features = 256-wide weighted scatter-add
__global__ void k_edge_agg(const int* __restrict__ src, const int* __restrict__ dst,
                           const float* __restrict__ w, const float* __restrict__ ssum,
                           const float* __restrict__ h, float* __restrict__ agg, int E) {
  int e = blockIdx.x * 8 + (threadIdx.x >> 5);
  if (e >= E) return;
  int lane = threadIdx.x & 31;
  int sj = src[e], di = dst[e];
  float alpha = w[e] / ssum[di];
  const float4* hp = (const float4*)(h + (size_t)sj * D_LAT) + lane * 2;
  float4 a = hp[0], b = hp[1];
  float* out = agg + (size_t)di * D_LAT + lane * 8;
  atomicAdd(out + 0, alpha * a.x); atomicAdd(out + 1, alpha * a.y);
  atomicAdd(out + 2, alpha * a.z); atomicAdd(out + 3, alpha * a.w);
  atomicAdd(out + 4, alpha * b.x); atomicAdd(out + 5, alpha * b.y);
  atomicAdd(out + 6, alpha * b.z); atomicAdd(out + 7, alpha * b.w);
}

// ---------------------------------------------------------------------------
extern "C" void kernel_launch(void* const* d_in, const int* in_sizes, int n_in,
                              void* d_out, int out_size, void* d_ws, size_t ws_size,
                              hipStream_t stream) {
  const float* x   = (const float*)d_in[0];
  const float* W1  = (const float*)d_in[1];
  const float* a1s = (const float*)d_in[2];
  const float* a1d = (const float*)d_in[3];
  const float* W2  = (const float*)d_in[4];
  const int* edges = (const int*)d_in[5];
  const int NN = in_sizes[0] / D_IN;   // 50000
  const int E  = in_sizes[5] / 2;      // 800000
  const int* src = edges;
  const int* dst = edges + E;

  float* emb   = (float*)d_out;                          // [NN, 64]
  float* recon = (float*)d_out + (size_t)NN * D_EMB;     // [NN, 512]

  // workspace carve-out (256B aligned slots)
  char* ws = (char*)d_ws; size_t off = 0;
  auto carve = [&](size_t bytes) -> void* {
    void* p = ws + off; off += (bytes + 255) & ~(size_t)255; return p;
  };
  float*    hA   = (float*)carve((size_t)NN * D_LAT * sizeof(float));  // h1, later h3 (fp32)
  float*    hB   = (float*)carve((size_t)NN * D_LAT * sizeof(float));  // agg buffers (fp32)
  bf16_t*   xb   = (bf16_t*)carve((size_t)NN * D_IN * sizeof(bf16_t)); // bf16(x)
  bf16_t*   latb = (bf16_t*)carve((size_t)NN * D_LAT * sizeof(bf16_t));// bf16 inter/dec_inter
  bf16_t*   embb = (bf16_t*)carve((size_t)NN * D_EMB * sizeof(bf16_t));// bf16(emb)
  bf16_t*   W1b  = (bf16_t*)carve((size_t)D_IN * D_LAT * sizeof(bf16_t)); // [512,256]
  bf16_t*   W1bT = (bf16_t*)carve((size_t)D_LAT * D_IN * sizeof(bf16_t)); // [256,512]
  bf16_t*   W2b  = (bf16_t*)carve((size_t)D_LAT * D_EMB * sizeof(bf16_t));// [256,64]
  bf16_t*   W2bT = (bf16_t*)carve((size_t)D_EMB * D_LAT * sizeof(bf16_t));// [64,256]
  float*    als  = (float*)carve((size_t)NN * sizeof(float));
  float*    ald  = (float*)carve((size_t)NN * sizeof(float));
  unsigned* menc = (unsigned*)carve((size_t)NN * sizeof(unsigned));
  float*    ssum = (float*)carve((size_t)NN * sizeof(float));
  float*    wexp = (float*)carve((size_t)E * sizeof(float));
  (void)ws_size; (void)n_in; (void)out_size;

  const int TB = 256;
  const long long NLAT = (long long)NN * D_LAT;

  // ---- one-time operand preparation (streaming, HBM/L2 bound, negligible)
  {
    long long n8 = (long long)NN * D_IN / 8;
    k_f32_to_bf16<<<(int)((n8 + TB - 1) / TB), TB, 0, stream>>>(x, xb, n8);
  }
  k_f32_to_bf16<<<(D_IN * D_LAT / 8 + TB - 1) / TB, TB, 0, stream>>>(W1, W1b, D_IN * D_LAT / 8);
  k_f32_to_bf16_T<<<(D_IN * D_LAT + TB - 1) / TB, TB, 0, stream>>>(W1, W1bT, D_IN, D_LAT);
  k_f32_to_bf16<<<(D_LAT * D_EMB / 8 + TB - 1) / TB, TB, 0, stream>>>(W2, W2b, D_LAT * D_EMB / 8);
  k_f32_to_bf16_T<<<(D_LAT * D_EMB + TB - 1) / TB, TB, 0, stream>>>(W2, W2bT, D_LAT, D_EMB);

  // ---- layer 1 GEMM: h1 = x @ W1   (A=xb, Bt=W1bT[N=256,K=512])
  {
    int tiles = (NN / 16) * (D_LAT / 16);
    k_gemm_wmma<<<(tiles + 7) / 8, TB, 0, stream>>>(
        xb, W1bT, hA, NN, D_LAT, D_IN, D_IN, D_IN, D_LAT);
  }
  // ---- per-node logits (shared by encoder & decoder attention — tied weights)
  k_logits<<<(NN + TB - 1) / TB, TB, 0, stream>>>(hA, a1s, a1d, als, ald, NN);

  // ---- segment softmax (computed ONCE, reused for both aggregations)
  k_fill_u32<<<(NN + TB - 1) / TB, TB, 0, stream>>>(menc, 0x007FFFFFu, NN); // enc(-inf)
  (void)hipMemsetAsync(ssum, 0, (size_t)NN * sizeof(float), stream);
  (void)hipMemsetAsync(hB, 0, (size_t)NLAT * sizeof(float), stream);
  k_edge_max<<<(E + TB - 1) / TB, TB, 0, stream>>>(src, dst, als, ald, menc, E);
  k_edge_expsum<<<(E + TB - 1) / TB, TB, 0, stream>>>(src, dst, als, ald, menc, wexp, ssum, E);

  // ---- encoder aggregation + fused ELU->bf16
  k_edge_agg<<<(E + 7) / 8, TB, 0, stream>>>(src, dst, wexp, ssum, hA, hB, E);
  k_elu_bf16<<<(int)((NLAT / 8 + TB - 1) / TB), TB, 0, stream>>>(hB, latb, NLAT / 8);

  // ---- emb = inter @ W2   (A=latb, Bt=W2bT[N=64,K=256]) -> d_out
  {
    int tiles = (NN / 16) * (D_EMB / 16);
    k_gemm_wmma<<<(tiles + 7) / 8, TB, 0, stream>>>(
        latb, W2bT, emb, NN, D_EMB, D_LAT, D_LAT, D_LAT, D_EMB);
  }
  // ---- h3 = emb @ W2^T    (A=bf16(emb), Bt=W2b[N=256,K=64])
  {
    long long n8 = (long long)NN * D_EMB / 8;
    k_f32_to_bf16<<<(int)((n8 + TB - 1) / TB), TB, 0, stream>>>(emb, embb, n8);
    int tiles = (NN / 16) * (D_LAT / 16);
    k_gemm_wmma<<<(tiles + 7) / 8, TB, 0, stream>>>(
        embb, W2b, hA, NN, D_LAT, D_EMB, D_EMB, D_EMB, D_LAT);
  }
  // ---- decoder aggregation (alpha3 == alpha1: reuse wexp/ssum) + fused ELU->bf16
  (void)hipMemsetAsync(hB, 0, (size_t)NLAT * sizeof(float), stream);
  k_edge_agg<<<(E + 7) / 8, TB, 0, stream>>>(src, dst, wexp, ssum, hA, hB, E);
  k_elu_bf16<<<(int)((NLAT / 8 + TB - 1) / TB), TB, 0, stream>>>(hB, latb, NLAT / 8);

  // ---- recon = dec_inter @ W1^T   (A=latb, Bt=W1b[N=512,K=256]) -> d_out
  {
    int tiles = (NN / 16) * (D_IN / 16);
    k_gemm_wmma<<<(tiles + 7) / 8, TB, 0, stream>>>(
        latb, W1b, recon, NN, D_IN, D_LAT, D_LAT, D_LAT, D_IN);
  }
}
